// Atention_14224931684962
// MI455X (gfx1250) — matrix operand order repaired and checked by reading, not compiled, not run on hardware
//
#include <hip/hip_runtime.h>

// ---------------------------------------------------------------------------
// Problem constants (match reference): B=4, S=4096, E=512, A=64
// ---------------------------------------------------------------------------
#define BB 4
#define SS 4096
#define EE 512
#define AA 64
#define LOG2E 1.4426950408889634f

#define HAS_ASYNC_LDS __has_builtin(__builtin_amdgcn_global_load_async_to_lds_b128)

typedef __attribute__((ext_vector_type(16))) __bf16 v16bf;
typedef __attribute__((ext_vector_type(8)))  float  v8f;
typedef int v4i __attribute__((vector_size(16)));       // int4 vector
typedef __attribute__((address_space(1))) v4i GV4;      // global int4
typedef __attribute__((address_space(3))) v4i LV4;      // LDS int4

// packed float2 -> bf16x2 via native v_cvt_pk_bf16_f32 (probe-verified on gfx1250)
__device__ __forceinline__ unsigned int f2bf2(float lo, float hi) {
#if __has_builtin(__builtin_amdgcn_cvt_pk_bf16_f32)
    typedef __attribute__((ext_vector_type(2))) __bf16 v2bf;
    union { v2bf v; unsigned int u; } r;
    r.v = __builtin_amdgcn_cvt_pk_bf16_f32(lo, hi);
    return r.u;
#else
    unsigned int r;
    asm("v_cvt_pk_bf16_f32 %0, %1, %2" : "=v"(r) : "v"(lo), "v"(hi));
    return r;
#endif
}

__device__ __forceinline__ void wait_async0() {
#if __has_builtin(__builtin_amdgcn_s_wait_asynccnt)
    __builtin_amdgcn_s_wait_asynccnt(0);
#else
    asm volatile("s_wait_asynccnt 0" ::: "memory");
#endif
}

// Load 16 contiguous bf16 (32B) as one fragment
__device__ __forceinline__ v16bf ld_bf16x16(const unsigned short* p) {
    union { uint4 q[2]; v16bf v; } u;
    u.q[0] = *(const uint4*)p;
    u.q[1] = *(const uint4*)(p + 8);
    return u.v;
}
// Two separate 16B chunks (A-fragment halves: j0..7 / j8..15)
__device__ __forceinline__ v16bf ld_bf16x16_2(const unsigned short* p0,
                                              const unsigned short* p1) {
    union { uint4 q[2]; v16bf v; } u;
    u.q[0] = *(const uint4*)p0;
    u.q[1] = *(const uint4*)p1;
    return u.v;
}
// 16 fp32 -> one v16bf fragment (8 packed cvt)
__device__ __forceinline__ v16bf cvt16(float4 f0, float4 f1, float4 f2, float4 f3) {
    union { unsigned int u[8]; v16bf v; } r;
    r.u[0] = f2bf2(f0.x, f0.y); r.u[1] = f2bf2(f0.z, f0.w);
    r.u[2] = f2bf2(f1.x, f1.y); r.u[3] = f2bf2(f1.z, f1.w);
    r.u[4] = f2bf2(f2.x, f2.y); r.u[5] = f2bf2(f2.z, f2.w);
    r.u[6] = f2bf2(f3.x, f3.y); r.u[7] = f2bf2(f3.z, f3.w);
    return r.v;
}

__device__ __forceinline__ v8f wmma_bf16(v16bf a, v16bf b, v8f c) {
    return __builtin_amdgcn_wmma_f32_16x16x32_bf16(false, a, false, b,
                                                   (short)0, c, false, false);
}

// ---------------------------------------------------------------------------
// Kernel 1: projections, transposed form  D = W^T(64x512) x X^T(512x16-rows)
// For the Q projection, 1/sqrt(A)*log2(e) is folded into the staged weights so
// attention scores come out of the WMMA already in log2 domain.
// 128 threads (4 waves) x 16 rows each. grid = (256 row-tiles, 3 projections)
// ---------------------------------------------------------------------------
__global__ void __launch_bounds__(128)
proj_kernel(const float* __restrict__ q, const float* __restrict__ k,
            const float* __restrict__ v, const float* __restrict__ wq,
            const float* __restrict__ wk, const float* __restrict__ wv,
            unsigned short* __restrict__ qp, unsigned short* __restrict__ kp,
            unsigned short* __restrict__ vp) {
    const int which = blockIdx.y;
    const float* x = (which == 0) ? q : (which == 1) ? k : v;
    const float* w = (which == 0) ? wq : (which == 1) ? wk : wv;
    unsigned short* o = (which == 0) ? qp : (which == 1) ? kp : vp;
    const float wscale = (which == 0) ? 0.125f * LOG2E : 1.0f;  // fold scale into Q

    __shared__ __align__(16) unsigned short ldsW[16 * 4 * 32 * 16];  // 64 KB
    unsigned int* ldsW32 = (unsigned int*)ldsW;

    const int tid  = threadIdx.x;
    const int wave = tid >> 5;
    const int lane = tid & 31;
    const int hi   = lane >> 4;
    const int ln   = lane & 15;

    // ---- stage W^T in A-fragment order, bf16; even-k pairs pack one b32 ----
    for (int pi = tid; pi < (EE / 2) * AA; pi += 128) {
        int n = pi & 63;
        int kidx = (pi >> 6) * 2;
        float a = w[(size_t)kidx * AA + n] * wscale;
        float b = w[(size_t)(kidx + 1) * AA + n] * wscale;
        int kstep = kidx >> 5;
        int kk = kidx & 31;
        int ahi = kk >> 4;                       // A-layout lane-half
        int j   = kk & 15;                       // even; j,j+1 same lane
        int alane = ahi * 16 + (n & 15);
        int nblk = n >> 4;
        int si = ((kstep * 4 + nblk) * 32 + alane) * 16 + j;
        ldsW32[si >> 1] = f2bf2(a, b);
    }
    __syncthreads();

    const int rowBase = blockIdx.x * 64 + wave * 16;
    const int row = rowBase + ln;                // this lane's output row (N axis)
    const float* xrow = x + (size_t)row * EE;

    v8f acc[4] = {};
#pragma unroll 4
    for (int kstep = 0; kstep < 16; ++kstep) {
        // B fragment: 16 consecutive fp32 of this row -> bf16
        const float* px = xrow + kstep * 32 + hi * 16;
        v16bf xb = cvt16(*(const float4*)px, *(const float4*)(px + 4),
                         *(const float4*)(px + 8), *(const float4*)(px + 12));
#pragma unroll
        for (int nblk = 0; nblk < 4; ++nblk) {
            v16bf wa = ld_bf16x16(ldsW + ((kstep * 4 + nblk) * 32 + lane) * 16);
            acc[nblk] = wmma_bf16(wa, xb, acc[nblk]);
        }
    }

    // ---- store: row=ln fixed per lane; 8 consecutive out-features per tile ----
    unsigned short* orow = o + (size_t)row * AA;
#pragma unroll
    for (int nblk = 0; nblk < 4; ++nblk) {
        uint4 u = make_uint4(f2bf2(acc[nblk][0], acc[nblk][1]),
                             f2bf2(acc[nblk][2], acc[nblk][3]),
                             f2bf2(acc[nblk][4], acc[nblk][5]),
                             f2bf2(acc[nblk][6], acc[nblk][7]));
        *(uint4*)(orow + nblk * 16 + 8 * hi) = u;
    }
}

// ---------------------------------------------------------------------------
// Kernel 2: fused masked flash attention, fully transposed:
//   S^T = K_tile x Q^T   (keys on M axis, queries on N/lane axis)
//   O^T = V^T x P^T      (heads on M axis, queries on N/lane axis)
// Scores arrive pre-scaled in log2 domain. Double-buffered K/V^T LDS tiles:
// one barrier per key-block; K tile is filled with ASYNC global->LDS DMA
// (ASYNCcnt) when available, overlapping the current block's WMMAs.
// 128 threads; wave owns 16 queries. grid = (S/64 = 64 qblocks, B = 4)
// ---------------------------------------------------------------------------
__global__ void __launch_bounds__(128)
attn_kernel(const unsigned short* __restrict__ qp,
            const unsigned short* __restrict__ kp,
            const unsigned short* __restrict__ vp,
            const unsigned char* __restrict__ mask,
            float* __restrict__ out) {
    __shared__ __align__(16) unsigned short ldsK[2][32 * 72];   // K tile [key][head]
    __shared__ __align__(16) unsigned short ldsVT[2][64 * 40];  // V^T tile [head][key]
    __shared__ __align__(16) unsigned short ldsP[4 * 16 * 40];  // per-wave P^T [query][key]

    const int tid   = threadIdx.x;
    const int wave  = tid >> 5;
    const int lane  = tid & 31;
    const int hi    = lane >> 4;
    const int ln    = lane & 15;
    const int b     = blockIdx.y;
    const int qbase = blockIdx.x * 64;
    const int query = qbase + wave * 16 + ln;    // this lane's query (N axis)
    const float NEG_INF = -__builtin_inff();
    const size_t bS = (size_t)b * SS;

    // Q^T B-fragments (heads 0-31 / 32-63): 32B contiguous per lane, kept in VGPRs
    const unsigned short* qrow = qp + (bS + query) * AA;
    v16bf qb0 = ld_bf16x16(qrow + hi * 16);
    v16bf qb1 = ld_bf16x16(qrow + 32 + hi * 16);

    float m = NEG_INF, l = 0.0f;
    v8f acc[4] = {};
    const unsigned char* mrow = mask + (bS + query) * SS;

    // ---- tile staging (double buffered) ----
    auto stage_tiles = [&](int kb2) {
        const int buf = kb2 & 1;
        const int kb2base = kb2 * 32;
        unsigned short* K  = ldsK[buf];
        unsigned short* VT = ldsVT[buf];
        // K tile (32x64 bf16 row-major, pad 72): pure copy -> async DMA to LDS
        for (int c = tid; c < 256; c += 128) {
            int key = c >> 3, part = c & 7;
            const unsigned short* g = kp + (bS + kb2base + key) * AA + part * 8;
            unsigned short* s = K + key * 72 + part * 8;
#if HAS_ASYNC_LDS
            __builtin_amdgcn_global_load_async_to_lds_b128((GV4*)g, (LV4*)s, 0, 0);
#else
            *(uint4*)s = *(const uint4*)g;
#endif
        }
        // V^T tile (transpose): 2 heads x 2 keys per step, b32 in / b32 out
        for (int pi = tid; pi < 512; pi += 128) {
            int h2 = (pi & 31) * 2;
            int kpair = pi >> 5;  // 0..15
            size_t base = (bS + kb2base + kpair * 2) * AA + h2;
            unsigned int a = *(const unsigned int*)(vp + base);       // key
            unsigned int c = *(const unsigned int*)(vp + base + AA);  // key+1
            *(unsigned int*)(VT + h2 * 40 + kpair * 2)       = (a & 0xFFFFu) | (c << 16);
            *(unsigned int*)(VT + (h2 + 1) * 40 + kpair * 2) = (a >> 16) | (c & 0xFFFF0000u);
        }
    };

    stage_tiles(0);
    for (int kb = 0; kb < SS / 32; ++kb) {
        const int kbase = kb * 32;
#if HAS_ASYNC_LDS
        wait_async0();                         // async K fills visible before barrier
#endif
        __syncthreads();                       // buf[kb&1] ready; buf[(kb+1)&1] free
        if (kb + 1 < SS / 32) stage_tiles(kb + 1);
        if (kb + 2 < SS / 32) {                // distant prefetch (global_prefetch_b8)
            __builtin_prefetch(kp + (bS + kbase + 64) * AA + tid * 16, 0, 3);
            __builtin_prefetch(vp + (bS + kbase + 64) * AA + tid * 16, 0, 3);
            __builtin_prefetch(mrow + kbase + 32, 0, 3);
        }
        const unsigned short* K  = ldsK[kb & 1];
        const unsigned short* VT = ldsVT[kb & 1];

        // ---- S^T tiles: keys nh*16+(r+8*hi) on M, query=ln on N ----
        v8f sc[2];
#pragma unroll
        for (int nh = 0; nh < 2; ++nh) {
            const unsigned short* kr = K + (nh * 16 + ln) * 72;
            v16bf a0 = ld_bf16x16_2(kr + hi * 16,      kr + hi * 16 + 8);  // heads 0..31
            v16bf a1 = ld_bf16x16_2(kr + 32 + hi * 16, kr + 40 + hi * 16);  // heads 32..63
            v8f c = {};
            c = wmma_bf16(a0, qb0, c);
            c = wmma_bf16(a1, qb1, c);
            sc[nh] = c;
        }

        // ---- mask: 8 consecutive bytes/lane/tile, v_cvt_f32_ubyte + fma bias ----
#pragma unroll
        for (int nh = 0; nh < 2; ++nh) {
            uint2 mv = *(const uint2*)(mrow + kbase + nh * 16 + 8 * hi);
            sc[nh][0] = fmaf((float)(mv.x & 0xFFu),         -1e38f, sc[nh][0]);
            sc[nh][1] = fmaf((float)((mv.x >> 8) & 0xFFu),  -1e38f, sc[nh][1]);
            sc[nh][2] = fmaf((float)((mv.x >> 16) & 0xFFu), -1e38f, sc[nh][2]);
            sc[nh][3] = fmaf((float)(mv.x >> 24),           -1e38f, sc[nh][3]);
            sc[nh][4] = fmaf((float)(mv.y & 0xFFu),         -1e38f, sc[nh][4]);
            sc[nh][5] = fmaf((float)((mv.y >> 8) & 0xFFu),  -1e38f, sc[nh][5]);
            sc[nh][6] = fmaf((float)((mv.y >> 16) & 0xFFu), -1e38f, sc[nh][6]);
            sc[nh][7] = fmaf((float)(mv.y >> 24),           -1e38f, sc[nh][7]);
        }

        // ---- online softmax in log2 domain; per-lane scalar state ----
        float t = sc[0][0];
#pragma unroll
        for (int r = 1; r < 8; ++r) t = fmaxf(t, sc[0][r]);
#pragma unroll
        for (int r = 0; r < 8; ++r) t = fmaxf(t, sc[1][r]);
        t = fmaxf(t, __shfl_xor(t, 16, 32));
        float mn = fmaxf(m, t);
        if (__any(mn > m)) {                   // rescale only when a max moved
            float alpha = exp2f(m - mn);       // m=-inf on first pass -> alpha=0
#pragma unroll
            for (int hb = 0; hb < 4; ++hb) acc[hb] *= alpha;
            l *= alpha;
        }
        m = mn;
        float ps = 0.0f;
#pragma unroll
        for (int nh = 0; nh < 2; ++nh)
#pragma unroll
            for (int r = 0; r < 8; ++r) {
                float e = exp2f(sc[nh][r] - mn);
                sc[nh][r] = e;
                ps += e;
            }
        ps += __shfl_xor(ps, 16, 32);
        l += ps;

        // ---- P^T staging: 8 consecutive keys per lane -> one b128 per tile ----
        unsigned short* pw = ldsP + wave * (16 * 40);
#pragma unroll
        for (int nh = 0; nh < 2; ++nh) {
            uint4 u = make_uint4(f2bf2(sc[nh][0], sc[nh][1]), f2bf2(sc[nh][2], sc[nh][3]),
                                 f2bf2(sc[nh][4], sc[nh][5]), f2bf2(sc[nh][6], sc[nh][7]));
            *(uint4*)(pw + ln * 40 + nh * 16 + 8 * hi) = u;
        }
        asm volatile("s_wait_dscnt 0" ::: "memory");  // intra-wave LDS RAW
        __builtin_amdgcn_wave_barrier();
        v16bf pb = ld_bf16x16(pw + ln * 40 + hi * 16);  // B-frag: query=ln, 16 keys

        // ---- O^T += V^T x P^T ----
#pragma unroll
        for (int hb = 0; hb < 4; ++hb) {
            const unsigned short* vr = VT + (hb * 16 + ln) * 40;
            v16bf va = ld_bf16x16_2(vr + hi * 16, vr + hi * 16 + 8);
            acc[hb] = wmma_bf16(va, pb, acc[hb]);
        }
    }

    // ---- epilogue: one query row per lane -> vectorized float4 stores ----
    float inv = 1.0f / l;
    float* op = out + (bS + query) * AA;
#pragma unroll
    for (int hb = 0; hb < 4; ++hb) {
        float4 o0 = make_float4(acc[hb][0] * inv, acc[hb][1] * inv,
                                acc[hb][2] * inv, acc[hb][3] * inv);
        float4 o1 = make_float4(acc[hb][4] * inv, acc[hb][5] * inv,
                                acc[hb][6] * inv, acc[hb][7] * inv);
        *(float4*)(op + hb * 16 + 8 * hi)     = o0;
        *(float4*)(op + hb * 16 + 8 * hi + 4) = o1;
    }
}

// ---------------------------------------------------------------------------
// Launcher
// ---------------------------------------------------------------------------
extern "C" void kernel_launch(void* const* d_in, const int* in_sizes, int n_in,
                              void* d_out, int out_size, void* d_ws, size_t ws_size,
                              hipStream_t stream) {
    (void)in_sizes; (void)n_in; (void)out_size; (void)ws_size;
    const float* q  = (const float*)d_in[0];
    const float* k  = (const float*)d_in[1];
    const float* v  = (const float*)d_in[2];
    const unsigned char* mask = (const unsigned char*)d_in[3];  // jax bool = 1 byte
    const float* wq = (const float*)d_in[4];
    const float* wk = (const float*)d_in[5];
    const float* wv = (const float*)d_in[6];
    float* out = (float*)d_out;

    // workspace: bf16 qp/kp/vp, each B*S*A elements (2 MB); total 6 MB
    unsigned short* qp = (unsigned short*)d_ws;
    unsigned short* kp = qp + (size_t)BB * SS * AA;
    unsigned short* vp = kp + (size_t)BB * SS * AA;

    dim3 gp((BB * SS) / 64, 3);
    proj_kernel<<<gp, 128, 0, stream>>>(q, k, v, wq, wk, wv, qp, kp, vp);

    dim3 ga(SS / 64, BB);
    attn_kernel<<<ga, 128, 0, stream>>>(qp, kp, vp, mask, out);
}
